// AttentionPool2d_14516989460764
// MI455X (gfx1250) — matrix operand order repaired and checked by reading, not compiled, not run on hardware
//
#include <hip/hip_runtime.h>

// ---------------- problem constants ----------------
constexpr int kE   = 1024;   // embed dim
constexpr int kT   = 1025;   // tokens (1024 spatial + mean token)
constexpr int kTP  = 1152;   // T padded to 9 tiles of 128
constexpr int kB   = 16;     // batch
constexpr int kH   = 16;     // heads
constexpr int kCH  = 64;     // channels per head

// ---------------- WMMA types ----------------
typedef __attribute__((ext_vector_type(16))) __bf16 v16bf_t;
typedef __attribute__((ext_vector_type(8)))  float  v8f_t;

union Frag32B { uint4 q[2]; v16bf_t v; };

__device__ __forceinline__ unsigned short f2bf(float f) {
    unsigned int u = __float_as_uint(f);
    unsigned int r = 0x7FFFu + ((u >> 16) & 1u);      // round-to-nearest-even
    return (unsigned short)((u + r) >> 16);
}

__device__ __forceinline__ float waveReduceSum(float v) {
#pragma unroll
    for (int m = 16; m >= 1; m >>= 1) v += __shfl_xor(v, m, 32);
    return v;
}

// LDS byte offset of a generic pointer to a __shared__ object
// (LDS offset is the low 32 bits of the generic address)
__device__ __forceinline__ unsigned int ldsOff(const void* p) {
    return (unsigned int)(unsigned long long)p;
}

// ---------------- 1) mean over spatial tokens ----------------
__global__ __launch_bounds__(256) void mean_kernel(const float* __restrict__ x,
                                                   float* __restrict__ mean) {
    int row  = blockIdx.x * 8 + (threadIdx.x >> 5);   // b*kE + c
    int lane = threadIdx.x & 31;
    const float* p = x + (size_t)row * 1024;
    float s = 0.f;
    for (int t = lane; t < 1024; t += 32) s += p[t];
    s = waveReduceSum(s);
    if (lane == 0) mean[row] = s * (1.0f / 1024.0f);
}

// ---------------- 2) w_qkv rows [1024,3072) -> bf16 ----------------
__global__ __launch_bounds__(256) void cvt_wkv_kernel(const float* __restrict__ w_qkv,
                                                      unsigned short* __restrict__ wkv) {
    int i = blockIdx.x * 256 + threadIdx.x;           // 0 .. 2*1024*1024
    wkv[i] = f2bf(w_qkv[(size_t)kE * kE + i]);
}

// ---------------- 3) augmented activations, transposed [b][t][c], bf16 ----------------
__global__ __launch_bounds__(256) void xaug_kernel(const float* __restrict__ x,
                                                   const float* __restrict__ pos,
                                                   const float* __restrict__ mean,
                                                   unsigned short* __restrict__ xaug) {
    __shared__ float tile[32][33];                    // padded, conflict-free transpose
    int b = blockIdx.z, cT = blockIdx.y, tT = blockIdx.x;
    int tx = threadIdx.x & 31, ty = threadIdx.x >> 5;
    int tG = tT * 32 + tx;
#pragma unroll
    for (int cc = ty; cc < 32; cc += 8) {
        int c = cT * 32 + cc;
        float v = 0.f;                                // zero-fill pad t in [1025,1152)
        if (tG < kT) {
            float xv = (tG == 0) ? mean[b * kE + c]
                                 : x[((size_t)b * kE + c) * 1024 + (tG - 1)];
            v = xv + pos[(size_t)c * kT + tG];
        }
        tile[cc][tx] = v;
    }
    __syncthreads();
#pragma unroll
    for (int tt = ty; tt < 32; tt += 8) {
        int t = tT * 32 + tt;
        int c = cT * 32 + tx;
        xaug[((size_t)b * kTP + t) * kE + c] = f2bf(tile[tx][tt]);
    }
}

// ---------------- 4) q at t=0 (mean token) ----------------
__global__ __launch_bounds__(256) void q_kernel(const float* __restrict__ w_qkv,
                                                const float* __restrict__ b_qkv,
                                                const float* __restrict__ mean,
                                                const float* __restrict__ pos,
                                                float* __restrict__ q) {
    int row  = blockIdx.x * 8 + (threadIdx.x >> 5);   // b*kE + o
    int lane = threadIdx.x & 31;
    int b = row >> 10, o = row & 1023;
    float s = 0.f;
    for (int c = lane; c < kE; c += 32)
        s += w_qkv[(size_t)o * kE + c] * (mean[b * kE + c] + pos[(size_t)c * kT]);
    s = waveReduceSum(s);
    if (lane == 0) q[row] = s + b_qkv[o];
}

// ---------------- 5) KV projection GEMM (bf16 WMMA, async-copy pipeline) ----------------
// C[b][o][t] = sum_c Wkv[o][c] * Xaug[b][t][c] + b_qkv[1024+o]
// block: 128x128 tile, 8 waves of 64x32 (4x2 WMMA frags), K step 32
// 3-stage LDS rotation fed by GLOBAL_LOAD_ASYNC_TO_LDS_B128 (ASYNCcnt tracked),
// exactly one s_barrier per K-step: prefetch at step i writes buf[(i+1)%3], whose
// last readers were at step i-2 — two barrier generations back; each wave does
// s_wait_asynccnt before signaling, so barrier arrival implies tile visibility.
__global__ __launch_bounds__(256) void kv_gemm_kernel(const unsigned short* __restrict__ A,  // [2048][1024] bf16
                                                      const unsigned short* __restrict__ Bm, // [kB][kTP][1024] bf16
                                                      const float* __restrict__ b_qkv,
                                                      float* __restrict__ C) {               // [kB][2048][kTP]
    __shared__ __align__(16) unsigned short As[3][128][32];   // 3 x 8 KB
    __shared__ __align__(16) unsigned short Bs[3][128][32];   // 3 x 8 KB
    const int b    = blockIdx.z;
    const int mT   = blockIdx.y;                      // 0..15
    const int nT   = blockIdx.x;                      // 0..8
    const int tid  = threadIdx.x;
    const int lane = tid & 31;
    const int wave = tid >> 5;
    const int wm   = wave >> 2;                       // 0..1  -> M offset 64*wm
    const int wn   = wave & 3;                        // 0..3  -> N offset 32*wn
    const int rih  = lane & 15;                       // row within 16-wide fragment
    const int hsel = lane >> 4;                       // which K-half this lane holds
    const int kA0  = hsel ? 8 : 0;                    // A: lanes0-15 K{0..7,16..23}, lanes16-31 K{8..15,24..31}
    const int kA1  = kA0 + 16;
    const int kB0  = hsel * 16;                       // B: same K order as A (uses kA0/kA1)

    // per-thread staging chunks: 2 x 16B for A, 2 x 16B for B per K-step
    const int chunk0 = tid * 2, chunk1 = tid * 2 + 1;
    const int r0 = chunk0 >> 2, c80 = (chunk0 & 3) * 8;
    const int r1 = chunk1 >> 2, c81 = (chunk1 & 3) * 8;

    const unsigned short* Ap = A  + (size_t)mT * 128 * kE;
    const unsigned short* Bp = Bm + ((size_t)b * kTP + (size_t)nT * 128) * kE;

    // issue 4 async 16B copies (global -> LDS) for K-step k0 into buffer `buf`
    auto issue = [&](int buf, int k0) {
        const unsigned short* ga0 = Ap + (size_t)r0 * kE + k0 + c80;
        const unsigned short* ga1 = Ap + (size_t)r1 * kE + k0 + c81;
        const unsigned short* gb0 = Bp + (size_t)r0 * kE + k0 + c80;
        const unsigned short* gb1 = Bp + (size_t)r1 * kE + k0 + c81;
        unsigned int la0 = ldsOff(&As[buf][r0][c80]);
        unsigned int la1 = ldsOff(&As[buf][r1][c81]);
        unsigned int lb0 = ldsOff(&Bs[buf][r0][c80]);
        unsigned int lb1 = ldsOff(&Bs[buf][r1][c81]);
        asm volatile("global_load_async_to_lds_b128 %0, %1, off" :: "v"(la0), "v"(ga0) : "memory");
        asm volatile("global_load_async_to_lds_b128 %0, %1, off" :: "v"(la1), "v"(ga1) : "memory");
        asm volatile("global_load_async_to_lds_b128 %0, %1, off" :: "v"(lb0), "v"(gb0) : "memory");
        asm volatile("global_load_async_to_lds_b128 %0, %1, off" :: "v"(lb1), "v"(gb1) : "memory");
    };

    v8f_t acc[4][2] = {};

    issue(0, 0);                                      // preamble prefetch
    int cur = 0;
    constexpr int NIT = kE / 32;                      // 32 K-steps

    for (int it = 0; it < NIT; ++it) {
        int nxt = cur + 1; if (nxt == 3) nxt = 0;
        if (it + 1 < NIT) {
            issue(nxt, (it + 1) * 32);
            asm volatile("s_wait_asynccnt 0x4" ::: "memory");   // this step's 4 copies done
        } else {
            asm volatile("s_wait_asynccnt 0x0" ::: "memory");
        }
        __syncthreads();                               // all waves' copies visible; prior readers done

        Frag32B af[4], bfrag[2];
#pragma unroll
        for (int mi = 0; mi < 4; ++mi) {
            const unsigned short* ar = &As[cur][wm * 64 + mi * 16 + rih][0];
            af[mi].q[0] = *(const uint4*)(ar + kA0);
            af[mi].q[1] = *(const uint4*)(ar + kA1);
        }
#pragma unroll
        for (int ni = 0; ni < 2; ++ni) {
            const unsigned short* br = &Bs[cur][wn * 32 + ni * 16 + rih][0];
            bfrag[ni].q[0] = *(const uint4*)(br + kA0);
            bfrag[ni].q[1] = *(const uint4*)(br + kA1);
        }

#pragma unroll
        for (int mi = 0; mi < 4; ++mi)
#pragma unroll
            for (int ni = 0; ni < 2; ++ni)
                acc[mi][ni] = __builtin_amdgcn_wmma_f32_16x16x32_bf16(
                    false, af[mi].v, false, bfrag[ni].v,
                    (short)0, acc[mi][ni], false, false);

        cur = nxt;
    }

    // epilogue: C layout lane n=lane&15 is column, VGPR r -> row r + 8*(lane>>4)
    const int mRow = hsel * 8;
#pragma unroll
    for (int mi = 0; mi < 4; ++mi) {
        int gmB = mT * 128 + wm * 64 + mi * 16 + mRow;
#pragma unroll
        for (int ni = 0; ni < 2; ++ni) {
            int gn = nT * 128 + wn * 32 + ni * 16 + rih;
            float* dst = C + ((size_t)b * 2048 + gmB) * kTP + gn;
#pragma unroll
            for (int r = 0; r < 8; ++r)
                dst[(size_t)r * kTP] = acc[mi][ni][r] + b_qkv[1024 + gmB + r];
        }
    }
}

// ---------------- 6) single-query attention per (b,h) ----------------
__global__ __launch_bounds__(256) void attn_kernel(const float* __restrict__ q,
                                                   const float* __restrict__ kv,
                                                   float* __restrict__ av) {
    __shared__ float sc[kT];
    __shared__ float red[256];
    __shared__ float qs[kCH];
    int h = blockIdx.x, b = blockIdx.y;
    int tid = threadIdx.x;
    if (tid < kCH) qs[tid] = q[b * kE + h * kCH + tid];
    __syncthreads();

    const float* K = kv + ((size_t)b * 2048 + h * kCH) * kTP;
#pragma unroll
    for (int i = 0; i < 5; ++i) {
        int s = tid + i * 256;
        if (s < kT) {
            float acc = 0.f;
#pragma unroll 8
            for (int c = 0; c < kCH; ++c) acc += qs[c] * K[(size_t)c * kTP + s];
            sc[s] = acc * 0.125f;                     // scale^2 = 1/sqrt(CH)
        }
    }
    __syncthreads();

    float m = -3.4e38f;
    for (int i = 0; i < 5; ++i) { int s = tid + i * 256; if (s < kT) m = fmaxf(m, sc[s]); }
    red[tid] = m; __syncthreads();
    for (int off = 128; off > 0; off >>= 1) {
        if (tid < off) red[tid] = fmaxf(red[tid], red[tid + off]);
        __syncthreads();
    }
    float mx = red[0];
    __syncthreads();

    float ps = 0.f;
    for (int i = 0; i < 5; ++i) {
        int s = tid + i * 256;
        if (s < kT) { float e = __expf(sc[s] - mx); sc[s] = e; ps += e; }
    }
    red[tid] = ps; __syncthreads();
    for (int off = 128; off > 0; off >>= 1) {
        if (tid < off) red[tid] += red[tid + off];
        __syncthreads();
    }
    float inv = 1.0f / red[0];
    __syncthreads();

    const float* V = kv + ((size_t)b * 2048 + 1024 + h * kCH) * kTP;
    int c = tid & 63, part = tid >> 6;
    float acc = 0.f;
    for (int s = part; s < kT; s += 4) acc += sc[s] * V[(size_t)c * kTP + s];
    red[tid] = acc; __syncthreads();
    if (part == 0)
        av[b * kE + h * kCH + c] =
            (red[c] + red[64 + c] + red[128 + c] + red[192 + c]) * inv;
}

// ---------------- 7) final projection, token 0 only ----------------
__global__ __launch_bounds__(256) void out_kernel(const float* __restrict__ w_c,
                                                  const float* __restrict__ b_c,
                                                  const float* __restrict__ av,
                                                  float* __restrict__ out) {
    int row  = blockIdx.x * 8 + (threadIdx.x >> 5);   // b*kE + o
    int lane = threadIdx.x & 31;
    int b = row >> 10, o = row & 1023;
    float s = 0.f;
    for (int c = lane; c < kE; c += 32) s += w_c[(size_t)o * kE + c] * av[b * kE + c];
    s = waveReduceSum(s);
    if (lane == 0) out[row] = s + b_c[o];
}

// ---------------- launch ----------------
extern "C" void kernel_launch(void* const* d_in, const int* in_sizes, int n_in,
                              void* d_out, int out_size, void* d_ws, size_t ws_size,
                              hipStream_t stream) {
    (void)in_sizes; (void)n_in; (void)out_size; (void)ws_size;
    const float* x       = (const float*)d_in[0];   // [16,1024,1024]
    const float* pos_emb = (const float*)d_in[1];   // [1024,1025]
    const float* w_qkv   = (const float*)d_in[2];   // [3072,1024]
    const float* b_qkv   = (const float*)d_in[3];   // [3072]
    const float* w_c     = (const float*)d_in[4];   // [1024,1024]
    const float* b_c     = (const float*)d_in[5];   // [1024]
    float* out = (float*)d_out;                     // [16,1024]

    // workspace layout (bytes): total ~193.1 MB
    char* ws = (char*)d_ws;
    float*          mean = (float*)(ws);                       //  65536
    float*          qv   = (float*)(ws + 65536);               //  65536
    float*          av   = (float*)(ws + 131072);              //  65536
    unsigned short* Wkv  = (unsigned short*)(ws + 196608);     //  4 MB   bf16 [2048][1024]
    unsigned short* Xaug = (unsigned short*)(ws + 4390912);    //  37.7MB bf16 [16][1152][1024]
    float*          KV   = (float*)(ws + 42139648);            // 151 MB  f32  [16][2048][1152]

    mean_kernel   <<<2048, 256, 0, stream>>>(x, mean);
    cvt_wkv_kernel<<<8192, 256, 0, stream>>>(w_qkv, Wkv);
    xaug_kernel   <<<dim3(36, 32, 16), 256, 0, stream>>>(x, pos_emb, mean, Xaug);
    q_kernel      <<<2048, 256, 0, stream>>>(w_qkv, b_qkv, mean, pos_emb, qv);
    kv_gemm_kernel<<<dim3(9, 16, 16), 256, 0, stream>>>(Wkv, Xaug, b_qkv, KV);
    attn_kernel   <<<dim3(16, 16), 256, 0, stream>>>(qv, KV, av);
    out_kernel    <<<2048, 256, 0, stream>>>(w_c, b_c, av, out);
}
